// RGNN_Model_28879360098971
// MI455X (gfx1250) — matrix operand order repaired and checked by reading, not compiled, not run on hardware
//
#include <hip/hip_runtime.h>

typedef __bf16 bf16_t;
typedef __attribute__((ext_vector_type(16))) __bf16 v16bf;
typedef __attribute__((ext_vector_type(8)))  __bf16 v8bf;
typedef __attribute__((ext_vector_type(4)))  __bf16 v4bf;
typedef __attribute__((ext_vector_type(8)))  float  v8f;

#define N_NODES 8192
#define N_EDGES 262144
#define N_IN    512
#define H1      512
#define H2      256

// ---------------------------------------------------------------------------
// f32 -> (hi, lo) bf16 split (optionally ReLU first).  a ~= hi + lo with
// ~2^-16 relative residual; used for bf16x3 WMMA GEMMs with f32 accumulate.
// Processes 4 elements/thread (float4 in, 8B bf16x4 out).
// ---------------------------------------------------------------------------
template <bool RELU>
__global__ __launch_bounds__(256) void split_bf16x2(
    const float* __restrict__ in, bf16_t* __restrict__ hi,
    bf16_t* __restrict__ lo, int n4)
{
    const int i = blockIdx.x * 256 + threadIdx.x;
    if (i >= n4) return;
    float4 x = reinterpret_cast<const float4*>(in)[i];
    if (RELU) {
        x.x = fmaxf(x.x, 0.f); x.y = fmaxf(x.y, 0.f);
        x.z = fmaxf(x.z, 0.f); x.w = fmaxf(x.w, 0.f);
    }
    const float xs[4] = {x.x, x.y, x.z, x.w};
    v4bf h, l;
    #pragma unroll
    for (int j = 0; j < 4; ++j) {
        const bf16_t hh = (bf16_t)xs[j];
        h[j] = hh;
        l[j] = (bf16_t)(xs[j] - (float)hh);
    }
    reinterpret_cast<v4bf*>(hi)[i] = h;
    reinterpret_cast<v4bf*>(lo)[i] = l;
}

// ---------------------------------------------------------------------------
// bf16x3 WMMA GEMM:  C[M,N] = (Ah+Al) @ (Bh+Bl) (+ epilogue), f32 accumulate,
// dropping the lo*lo term:  C = Ah Bh + Ah Bl + Al Bh  (v_wmma_f32_16x16x32_bf16)
//   A*: [M,K] bf16 row-major
//   B*: [K,N] bf16 row-major, or if TRANS_B: [N,K] row-major (logical
//       B[k][n] = Bmat[n][k]; used for z @ z^T, no transpose needed at staging)
//   EPI == 0: C += bias[n]            (regular store)
//   EPI == 1: C[i][j] *= inv[i]*inv[j], diag zeroed  (nontemporal store: the
//             256MB S matrix exceeds the 192MB L2 and is never re-read)
// Block: 256 threads (8 wave32), 64x64 macro-tile, BK=32 (one WMMA K-step).
// Each wave owns a 16x32 strip: 6 WMMAs per k-chunk sharing the A fragments.
// LDS B panels are stored column-major [n][k] so each lane's B fragment is
// two runs of 8 consecutive K (two 16B ds_load_b128), matching the ISA 16-bit B layout:
// lanes 0-15 hold K=0..7,16..23, lanes 16-31 hold K=8..15,24..31.
// M,N multiples of 64; K multiple of 32 (all shapes here qualify).
// ---------------------------------------------------------------------------
template <bool TRANS_B, int EPI>
__global__ __launch_bounds__(256) void gemm_bf16x3(
    const bf16_t* __restrict__ Ah, const bf16_t* __restrict__ Al,
    const bf16_t* __restrict__ Bh, const bf16_t* __restrict__ Bl,
    float* __restrict__ C, const float* __restrict__ bias_or_inv,
    int M, int N, int K)
{
    // stride 40 bf16 = 80B: multiple of 16B (aligned b128) and 20 dwords
    // (16 lanes * 20 dwords mod 64 covers all banks -> conflict-free frags)
    __shared__ bf16_t la_h[64][40];
    __shared__ bf16_t la_l[64][40];
    __shared__ bf16_t lb_h[64][40];   // column-major: [n][k]
    __shared__ bf16_t lb_l[64][40];

    const int tid  = threadIdx.x;
    const int lane = tid & 31;
    const int wave = tid >> 5;
    const int half = lane >> 4;   // 0: lanes 0-15, 1: lanes 16-31
    const int l16  = lane & 15;

    const int blockM = blockIdx.y * 64;
    const int blockN = blockIdx.x * 64;

    // wave sub-tile: rows [sm, sm+16), cols [sn, sn+32)
    const int sm = (wave >> 1) * 16;
    const int sn = (wave & 1) * 32;

    v8f c0 = {0.f, 0.f, 0.f, 0.f, 0.f, 0.f, 0.f, 0.f};
    v8f c1 = {0.f, 0.f, 0.f, 0.f, 0.f, 0.f, 0.f, 0.f};

    // staging coordinates: 64 rows x 32 k, one v8bf (16B) per thread per panel
    const int ar  = tid >> 2;        // 0..63
    const int ak8 = (tid & 3) * 8;   // 0,8,16,24
    const int bk  = tid >> 3;        // 0..31   (!TRANS_B transpose staging)
    const int bn8 = (tid & 7) * 8;

    for (int kb = 0; kb < K; kb += 32) {
        // ---- stage A panels 64x32 ----
        {
            const size_t g = (size_t)(blockM + ar) * K + kb + ak8;
            v8bf vh = *reinterpret_cast<const v8bf*>(&Ah[g]);
            v8bf vl = *reinterpret_cast<const v8bf*>(&Al[g]);
            *reinterpret_cast<v8bf*>(&la_h[ar][ak8]) = vh;
            *reinterpret_cast<v8bf*>(&la_l[ar][ak8]) = vl;
        }
        // ---- stage B panels into [n][k] ----
        if (TRANS_B) {
            // Bmat rows are the needed columns: direct 16B row copies
            const size_t g = (size_t)(blockN + ar) * K + kb + ak8;
            v8bf vh = *reinterpret_cast<const v8bf*>(&Bh[g]);
            v8bf vl = *reinterpret_cast<const v8bf*>(&Bl[g]);
            *reinterpret_cast<v8bf*>(&lb_h[ar][ak8]) = vh;
            *reinterpret_cast<v8bf*>(&lb_l[ar][ak8]) = vl;
        } else {
            // row-major [K,N]: coalesced 16B reads, transposed scalar stores
            const size_t g = (size_t)(kb + bk) * N + blockN + bn8;
            v8bf vh = *reinterpret_cast<const v8bf*>(&Bh[g]);
            v8bf vl = *reinterpret_cast<const v8bf*>(&Bl[g]);
            #pragma unroll
            for (int j = 0; j < 8; ++j) {
                lb_h[bn8 + j][bk] = vh[j];
                lb_l[bn8 + j][bk] = vl[j];
            }
        }
        __syncthreads();

        // ---- fragments ----
        // A 16x32 bf16 layout: lane(m=l16, h): VGPR0-3 = K[8h..8h+8),
        // VGPR4-7 = K[16+8h..+8)  -> two contiguous 16B chunks per matrix
        const bf16_t* pah = &la_h[sm + l16][8 * half];
        const bf16_t* pal = &la_l[sm + l16][8 * half];
        v16bf a_h = __builtin_shufflevector(
            *(const v8bf*)pah, *(const v8bf*)(pah + 16),
            0,1,2,3,4,5,6,7,8,9,10,11,12,13,14,15);
        v16bf a_l = __builtin_shufflevector(
            *(const v8bf*)pal, *(const v8bf*)(pal + 16),
            0,1,2,3,4,5,6,7,8,9,10,11,12,13,14,15);
        // B 32x16 bf16 layout: lane(n=l16, h) holds K[8h..8h+8) then K[16+8h..+8)
        const bf16_t* pb0h = &lb_h[sn + l16][8 * half];
        const bf16_t* pb0l = &lb_l[sn + l16][8 * half];
        const bf16_t* pb1h = &lb_h[sn + 16 + l16][8 * half];
        const bf16_t* pb1l = &lb_l[sn + 16 + l16][8 * half];
        v16bf b_h0 = __builtin_shufflevector(
            *(const v8bf*)pb0h, *(const v8bf*)(pb0h + 16),
            0,1,2,3,4,5,6,7,8,9,10,11,12,13,14,15);
        v16bf b_l0 = __builtin_shufflevector(
            *(const v8bf*)pb0l, *(const v8bf*)(pb0l + 16),
            0,1,2,3,4,5,6,7,8,9,10,11,12,13,14,15);
        v16bf b_h1 = __builtin_shufflevector(
            *(const v8bf*)pb1h, *(const v8bf*)(pb1h + 16),
            0,1,2,3,4,5,6,7,8,9,10,11,12,13,14,15);
        v16bf b_l1 = __builtin_shufflevector(
            *(const v8bf*)pb1l, *(const v8bf*)(pb1l + 16),
            0,1,2,3,4,5,6,7,8,9,10,11,12,13,14,15);

        c0 = __builtin_amdgcn_wmma_f32_16x16x32_bf16(
            false, a_h, false, b_h0, (short)0, c0, false, false);
        c0 = __builtin_amdgcn_wmma_f32_16x16x32_bf16(
            false, a_h, false, b_l0, (short)0, c0, false, false);
        c0 = __builtin_amdgcn_wmma_f32_16x16x32_bf16(
            false, a_l, false, b_h0, (short)0, c0, false, false);
        c1 = __builtin_amdgcn_wmma_f32_16x16x32_bf16(
            false, a_h, false, b_h1, (short)0, c1, false, false);
        c1 = __builtin_amdgcn_wmma_f32_16x16x32_bf16(
            false, a_h, false, b_l1, (short)0, c1, false, false);
        c1 = __builtin_amdgcn_wmma_f32_16x16x32_bf16(
            false, a_l, false, b_h1, (short)0, c1, false, false);
        __syncthreads();
    }

    // ---- epilogue: C/D layout => VGPR r holds M = r + 8*half, N = l16 ----
    #pragma unroll
    for (int r = 0; r < 8; ++r) {
        const int row  = blockM + sm + r + 8 * half;
        const int col0 = blockN + sn + l16;
        const int col1 = col0 + 16;
        float v0 = c0[r], v1 = c1[r];
        if (EPI == 0) {
            v0 += bias_or_inv[col0];
            v1 += bias_or_inv[col1];
            C[(size_t)row * N + col0] = v0;
            C[(size_t)row * N + col1] = v1;
        } else {
            const float im = bias_or_inv[row];
            v0 *= im * bias_or_inv[col0];
            v1 *= im * bias_or_inv[col1];
            if (row == col0) v0 = 0.f;
            if (row == col1) v1 = 0.f;
            __builtin_nontemporal_store(v0, &C[(size_t)row * N + col0]);
            __builtin_nontemporal_store(v1, &C[(size_t)row * N + col1]);
        }
    }
}

// ---------------------------------------------------------------------------
// inv[i] = 1 / ||z_i||  (one wave32 per row, shuffle reduction)
// ---------------------------------------------------------------------------
__global__ __launch_bounds__(256) void rownorm_inv(
    const float* __restrict__ z, float* __restrict__ inv)
{
    const int row  = blockIdx.x * 8 + (threadIdx.x >> 5);
    const int lane = threadIdx.x & 31;
    const float* p = z + (size_t)row * N_IN;
    float s = 0.f;
    #pragma unroll
    for (int i = 0; i < N_IN / 32; ++i) {
        float v = p[lane + i * 32];
        s += v * v;
    }
    #pragma unroll
    for (int off = 16; off > 0; off >>= 1)
        s += __shfl_xor(s, off, 32);
    if (lane == 0) inv[row] = rsqrtf(s);
}

// ---------------------------------------------------------------------------
// out[dst] += w * h[src]  — one block per edge, global_atomic_add_f32 scatter
// ---------------------------------------------------------------------------
template <int F>
__global__ __launch_bounds__(256) void spmm_scatter(
    const int* __restrict__ src, const int* __restrict__ dst,
    const float* __restrict__ ew, const float* __restrict__ h,
    float* __restrict__ out)
{
    const int   e = blockIdx.x;
    const int   s = src[e];
    const int   d = dst[e];
    const float w = ew[e];
    const float* hp = h + (size_t)s * F;
    float*       op = out + (size_t)d * F;
    #pragma unroll
    for (int i = 0; i < F / 256; ++i) {
        const int f = threadIdx.x + i * 256;
        unsafeAtomicAdd(&op[f], w * hp[f]);
    }
}

__global__ __launch_bounds__(256) void zero_f32(float* __restrict__ p, size_t n)
{
    size_t i = (size_t)blockIdx.x * 256 + threadIdx.x;
    if (i < n) p[i] = 0.f;
}

// ---------------------------------------------------------------------------
extern "C" void kernel_launch(void* const* d_in, const int* in_sizes, int n_in,
                              void* d_out, int out_size, void* d_ws, size_t ws_size,
                              hipStream_t stream)
{
    const float* X     = (const float*)d_in[0];
    const int*   src   = (const int*)d_in[1];
    const int*   dst   = (const int*)d_in[2];
    const float* ew    = (const float*)d_in[3];
    const float* W_rev = (const float*)d_in[4];
    const float* b_rev = (const float*)d_in[5];
    const float* W1    = (const float*)d_in[6];
    const float* b1    = (const float*)d_in[7];
    const float* W2    = (const float*)d_in[8];
    const float* b2    = (const float*)d_in[9];

    float* out_h = (float*)d_out;                      // [8192, 256]
    float* out_S = out_h + (size_t)N_NODES * H2;       // [8192, 8192]

    const size_t NX  = (size_t)N_NODES * N_IN;         // 4194304
    const size_t NW  = (size_t)N_IN * H1;              // 262144
    const size_t NW2 = (size_t)H1 * H2;                // 131072

    // ---- workspace carve (~50.5 MB), regions reused by lifetime ----
    char* ws = (char*)d_ws;
    size_t off = 0;
    bf16_t* Xh  = (bf16_t*)(ws + off); off += NX  * 2;
    bf16_t* Xl  = (bf16_t*)(ws + off); off += NX  * 2;
    bf16_t* Wrh = (bf16_t*)(ws + off); off += NW  * 2;
    bf16_t* Wrl = (bf16_t*)(ws + off); off += NW  * 2;
    bf16_t* W1h = (bf16_t*)(ws + off); off += NW  * 2;
    bf16_t* W1l = (bf16_t*)(ws + off); off += NW  * 2;
    bf16_t* W2h = (bf16_t*)(ws + off); off += NW2 * 2;
    bf16_t* W2l = (bf16_t*)(ws + off); off += NW2 * 2;
    char* R2 = ws + off; off += NX * 4;   // z f32 -> t1 f32 -> (gh, gl)
    char* R3 = ws + off; off += NX * 4;   // (zh, zl) -> h1 f32 -> t2 f32
    float* inv = (float*)(ws + off); off += N_NODES * 4;

    float*  z  = (float*)R2;
    float*  t1 = (float*)R2;
    bf16_t* gh = (bf16_t*)R2;             // relu(h1) split
    bf16_t* gl = gh + NX;
    bf16_t* zh = (bf16_t*)R3;
    bf16_t* zl = zh + NX;
    float*  h1 = (float*)R3;
    float*  t2 = (float*)R3;

    const dim3 blk(256);
    const int  NX4 = (int)(NX / 4);

    // 1. split X and weights to bf16 hi/lo
    split_bf16x2<false><<<NX4 / 256, blk, 0, stream>>>(X, Xh, Xl, NX4);
    split_bf16x2<false><<<(int)(NW / 4) / 256, blk, 0, stream>>>(W_rev, Wrh, Wrl, (int)(NW / 4));
    split_bf16x2<false><<<(int)(NW / 4) / 256, blk, 0, stream>>>(W1, W1h, W1l, (int)(NW / 4));
    split_bf16x2<false><<<(int)(NW2 / 4) / 256, blk, 0, stream>>>(W2, W2h, W2l, (int)(NW2 / 4));

    // 2. z = X @ W_rev + b_rev
    gemm_bf16x3<false, 0><<<dim3(H1 / 64, N_NODES / 64), blk, 0, stream>>>(
        Xh, Xl, Wrh, Wrl, z, b_rev, N_NODES, H1, N_IN);
    // 3. split z; inverse row norms
    split_bf16x2<false><<<NX4 / 256, blk, 0, stream>>>(z, zh, zl, NX4);
    rownorm_inv<<<N_NODES / 8, blk, 0, stream>>>(z, inv);
    // 4. S = normalized z @ z^T, diag zeroed  (dominant: 68.7 GFLOP)
    gemm_bf16x3<true, 1><<<dim3(N_NODES / 64, N_NODES / 64), blk, 0, stream>>>(
        zh, zl, zh, zl, out_S, inv, N_NODES, N_NODES, H1);

    // 5. t1 = X @ W1 + b1
    gemm_bf16x3<false, 0><<<dim3(H1 / 64, N_NODES / 64), blk, 0, stream>>>(
        Xh, Xl, W1h, W1l, t1, b1, N_NODES, H1, N_IN);
    // 6. h1 = spmm(A, t1)
    const size_t n1 = NX;
    zero_f32<<<(unsigned)((n1 + 255) / 256), blk, 0, stream>>>(h1, n1);
    spmm_scatter<H1><<<N_EDGES, blk, 0, stream>>>(src, dst, ew, t1, h1);
    // 7. t2 = relu(h1) @ W2 + b2  (ReLU fused into the split)
    split_bf16x2<true><<<NX4 / 256, blk, 0, stream>>>(h1, gh, gl, NX4);
    gemm_bf16x3<false, 0><<<dim3(H2 / 64, N_NODES / 64), blk, 0, stream>>>(
        gh, gl, W2h, W2l, t2, b2, N_NODES, H2, H1);
    // 8. out_h = spmm(A, t2)
    const size_t n2 = (size_t)N_NODES * H2;
    zero_f32<<<(unsigned)((n2 + 255) / 256), blk, 0, stream>>>(out_h, n2);
    spmm_scatter<H2><<<N_EDGES, blk, 0, stream>>>(src, dst, ew, t2, out_h);
}
